// GRU_81235011436734
// MI455X (gfx1250) — matrix operand order repaired and checked by reading, not compiled, not run on hardware
//
#include <hip/hip_runtime.h>
#include <hip/hip_bf16.h>

// ---------------------------------------------------------------------------
// 2-layer GRU for MI455X (gfx1250, wave32, WMMA bf16)
//   T=512, B=64, D=256, H=512, gates r,z,n (PyTorch order)
// Phase 1: convert x + all weights to bf16, zero h state + barrier counters.
// Phase 2: xg = x @ W_ih^T + b_ih (parallel WMMA GEMM, fp32 out)
// Phase 3: persistent 16-WG recurrence; W_hh slice resident in LDS, h staged
//          per step via async global->LDS (ASYNCcnt), xg prefetched under the
//          GEMM, WMMA from LDS, fp32 gate math, device-scope step barrier.
// ---------------------------------------------------------------------------

typedef __bf16 bf16;
typedef __attribute__((ext_vector_type(16))) __bf16 v16bf;
typedef __attribute__((ext_vector_type(8)))  __bf16 bf16x8;
typedef __attribute__((ext_vector_type(8)))  float  v8f;

#define T_STEPS 512
#define BATCH   64
#define DIN     256
#define HID     512
#define N3      1536      // 3*H
#define NWG     16        // persistent workgroups for recurrence
#define HC      (HID/NWG) // 32 h-columns per workgroup
#define LDW     (HID + 8) // padded LDS row stride (halves) to spread banks

// ---- WMMA fragment loaders (CDNA5 16-bit layouts, CDNA5 ISA) ----
// A (16x32 bf16, MxK): lane L holds row M=L&15; lanes 0-15 carry K in
// {kb..kb+7} U {kb+16..kb+23} with kb=(L>>4)*8 -> two 16B contiguous loads.
__device__ inline v16bf load_a_frag(const bf16* __restrict__ base, int lane, int ldk) {
    const int row = lane & 15;
    const int kb  = (lane >> 4) * 8;
    union { v16bf v; bf16x8 h[2]; } u;
    const bf16* p = base + (size_t)row * ldk + kb;
    u.h[0] = *reinterpret_cast<const bf16x8*>(p);
    u.h[1] = *reinterpret_cast<const bf16x8*>(p + 16);
    return u.v;
}

// B (32x16 bf16, KxN): lane L holds col N=L&15; K in {kb..kb+7} U {kb+16..kb+23},
// kb=(L>>4)*8 (same as A). Weights stored [N][K] row-major -> two 16B loads/lane.
__device__ inline v16bf load_b_frag(const bf16* __restrict__ base, int lane, int ldk) {
    const int col = lane & 15;
    const int kb  = (lane >> 4) * 8;
    union { v16bf v; bf16x8 h[2]; } u;
    const bf16* p = base + (size_t)col * ldk + kb;
    u.h[0] = *reinterpret_cast<const bf16x8*>(p);
    u.h[1] = *reinterpret_cast<const bf16x8*>(p + 16);
    return u.v;
}

__device__ inline v8f wmma_bf16(v16bf a, v16bf b, v8f c) {
    return __builtin_amdgcn_wmma_f32_16x16x32_bf16(false, a, false, b,
                                                   (short)0, c, false, false);
}

// Async global->LDS copy, 16B per lane (GLOBAL_LOAD_ASYNC_TO_LDS_B128).
__device__ inline void async_g2l_b128(unsigned lds_off, const void* gptr) {
    asm volatile("global_load_async_to_lds_b128 %0, %1, off"
                 :: "v"(lds_off), "v"(gptr) : "memory");
}
__device__ inline void wait_async0() {
    asm volatile("s_wait_asynccnt 0x0" ::: "memory");
}

// ---- Phase 1: fp32 -> bf16 conversion + state/counter init --------------
__global__ __launch_bounds__(256)
void gru_init(const float* __restrict__ x,
              const float* __restrict__ wih0, const float* __restrict__ whh0,
              const float* __restrict__ wih1, const float* __restrict__ whh1,
              bf16* __restrict__ xb,
              bf16* __restrict__ bwih0, bf16* __restrict__ bwhh0,
              bf16* __restrict__ bwih1, bf16* __restrict__ bwhh1,
              bf16* __restrict__ hb0, bf16* __restrict__ hb1,
              unsigned* __restrict__ ctrs) {
    const size_t i      = (size_t)blockIdx.x * blockDim.x + threadIdx.x;
    const size_t stride = (size_t)gridDim.x * blockDim.x;
    const size_t nxb   = (size_t)T_STEPS * BATCH * DIN;
    const size_t nwih0 = (size_t)N3 * DIN;
    const size_t nwhh  = (size_t)N3 * HID;
    const size_t nh    = (size_t)2 * BATCH * HID;
    for (size_t k = i; k < nxb;   k += stride) xb[k]    = (bf16)x[k];
    for (size_t k = i; k < nwih0; k += stride) bwih0[k] = (bf16)wih0[k];
    for (size_t k = i; k < nwhh;  k += stride) {
        bwhh0[k] = (bf16)whh0[k];
        bwih1[k] = (bf16)wih1[k];
        bwhh1[k] = (bf16)whh1[k];
    }
    for (size_t k = i; k < nh; k += stride) { hb0[k] = (bf16)0.0f; hb1[k] = (bf16)0.0f; }
    if (i < 8) ctrs[i] = 0u;
}

// ---- Phase 2: input projection GEMM  xg[M,N3] = A[M,K] @ W[N3,K]^T + b ---
// Block = 256 threads (8 waves), block tile 64x128, wave tile 32x32 (2x2 WMMA).
__global__ __launch_bounds__(256)
void gru_xproj(const bf16* __restrict__ A, const bf16* __restrict__ W,
               const float* __restrict__ bias, float* __restrict__ xg, int K) {
    const int lane  = threadIdx.x & 31;
    const int wave  = threadIdx.x >> 5;
    const int waveM = wave >> 2;          // 0..1
    const int waveN = wave & 3;           // 0..3
    const int m0 = blockIdx.y * 64 + waveM * 32;
    const int n0 = blockIdx.x * 128 + waveN * 32;

    v8f acc[2][2] = {};
    for (int kc = 0; kc < K; kc += 32) {
        v16bf a0 = load_a_frag(A + (size_t)m0 * K + kc,        lane, K);
        v16bf a1 = load_a_frag(A + (size_t)(m0 + 16) * K + kc, lane, K);
        v16bf b0 = load_b_frag(W + (size_t)n0 * K + kc,        lane, K);
        v16bf b1 = load_b_frag(W + (size_t)(n0 + 16) * K + kc, lane, K);
        acc[0][0] = wmma_bf16(a0, b0, acc[0][0]);
        acc[0][1] = wmma_bf16(a0, b1, acc[0][1]);
        acc[1][0] = wmma_bf16(a1, b0, acc[1][0]);
        acc[1][1] = wmma_bf16(a1, b1, acc[1][1]);
    }
    #pragma unroll
    for (int tm = 0; tm < 2; ++tm) {
        #pragma unroll
        for (int tn = 0; tn < 2; ++tn) {
            const int n  = n0 + tn * 16 + (lane & 15);
            const float bv = bias[n];
            #pragma unroll
            for (int v = 0; v < 8; ++v) {
                const int m = m0 + tm * 16 + v + ((lane >> 4) << 3);
                xg[(size_t)m * N3 + n] = acc[tm][tn][v] + bv;
            }
        }
    }
}

// ---- Phase 3: persistent recurrence ------------------------------------
// NWG workgroups x 256 threads. WG `cid` owns h columns [cid*HC, cid*HC+HC).
// LDS: W_hh slice (96 rows, resident) + per-step h stage + gate buffer.
__global__ __launch_bounds__(256)
void gru_recur(const float* __restrict__ xg,   // [T*B, 3H] fp32
               const bf16*  __restrict__ whh,  // [3H, H] bf16
               const float* __restrict__ bhh,  // [3H]
               bf16*        __restrict__ hb,   // [2, B, H] bf16 (double buffer)
               bf16*        __restrict__ ybf,  // layer0 out (bf16) or nullptr
               float*       __restrict__ yf,   // layer1 out (fp32) or nullptr
               float*       __restrict__ hn_out, // [B, H] final hidden
               unsigned*    __restrict__ ctr) {
    __shared__ bf16  wslice[3 * HC * LDW];      // 97.5 KB: this WG's W_hh rows
    __shared__ bf16  hstage[BATCH * LDW];       // 65 KB: bf16 h, staged per step
    __shared__ float gatebuf[3][BATCH][HC];     // 24 KB: r/z/n pre-activations

    const int tid  = threadIdx.x;
    const int lane = tid & 31;
    const int wave = tid >> 5;
    const int c0   = blockIdx.x * HC;

    // ---- one-time: stage this WG's 96 W_hh rows into LDS ----
    for (int i = tid; i < 3 * HC * (HID / 8); i += 256) {   // 16B chunks
        const int rl = i >> 6;        // 0..95
        const int ck = i & 63;        // 16B chunk within row
        const int g  = rl >> 5;
        const int nl = rl & 31;
        const bf16* src = whh + (size_t)(g * HID + c0 + nl) * HID + ck * 8;
        *reinterpret_cast<bf16x8*>(&wslice[rl * LDW + ck * 8]) =
            *reinterpret_cast<const bf16x8*>(src);
    }

    // fp32 master copy of this thread's 8 fixed h elements
    float hreg[8];
    #pragma unroll
    for (int l = 0; l < 8; ++l) hreg[l] = 0.0f;
    __syncthreads();

    const unsigned hstage_base = (unsigned)(size_t)&hstage[0];

    for (int t = 0; t < T_STEPS; ++t) {
        const bf16* hsrc = hb + (size_t)(t & 1) * BATCH * HID;
        bf16*       hdst = hb + (size_t)((t + 1) & 1) * BATCH * HID;

        // ---- async-stage h[t] (64KB) from L2 into LDS ----
        for (int i = tid; i < BATCH * (HID / 8); i += 256) { // 4096 16B chunks
            const int r  = i >> 6;    // batch row
            const int ck = i & 63;    // 16B chunk
            async_g2l_b128(hstage_base + (unsigned)(r * LDW + ck * 8) * 2,
                           hsrc + (size_t)r * HID + ck * 8);
        }

        // ---- prefetch this step's xg gate slices; they land under the GEMM.
        // One lane per wave: covers all 64 batch rows x 3 gates (128B each).
        if ((tid & 31) == 0) {
            #pragma unroll
            for (int l = 0; l < 8; ++l) {
                const int b = (tid >> 5) + l * 8;          // batch row
                const float* base = xg + ((size_t)t * BATCH + b) * N3 + c0;
                __builtin_prefetch(base, 0, 3);
                __builtin_prefetch(base + HID, 0, 3);
                __builtin_prefetch(base + 2 * HID, 0, 3);
            }
        }

        wait_async0();
        __syncthreads();

        // ---- hidden-state GEMM from LDS: 3 tiles per wave ----
        #pragma unroll
        for (int j = 0; j < 3; ++j) {
            const int tile = wave * 3 + j;           // 0..23
            const int g    = tile >> 3;              // gate 0..2
            const int mt   = (tile & 7) >> 1;        // 0..3
            const int nt   = tile & 1;               // 0..1
            const int m0   = mt * 16;
            v8f acc = {};
            for (int kc = 0; kc < HID; kc += 32) {
                v16bf a = load_a_frag(&hstage[m0 * LDW + kc], lane, LDW);
                v16bf b = load_b_frag(&wslice[(g * HC + nt * 16) * LDW + kc], lane, LDW);
                acc = wmma_bf16(a, b, acc);
            }
            const int nl = nt * 16 + (lane & 15);    // local col in gate chunk
            const float bv = bhh[g * HID + c0 + nl];
            #pragma unroll
            for (int v = 0; v < 8; ++v) {
                const int m = m0 + v + ((lane >> 4) << 3);
                gatebuf[g][m][nl] = acc[v] + bv;
            }
        }
        __syncthreads();

        // ---- gate nonlinearity + blend (fp32); fixed idx -> h in registers ----
        #pragma unroll
        for (int l = 0; l < (BATCH * HC) / 256; ++l) {
            const int idx = tid + l * 256;
            const int b   = idx >> 5;       // HC == 32
            const int cl  = idx & 31;
            const int c   = c0 + cl;
            const size_t row = (size_t)t * BATCH + b;
            const float xr = xg[row * N3 + c];
            const float xz = xg[row * N3 + HID + c];
            const float xn = xg[row * N3 + 2 * HID + c];
            const float hr = gatebuf[0][b][cl];
            const float hz = gatebuf[1][b][cl];
            const float hn = gatebuf[2][b][cl];
            const float r  = 1.0f / (1.0f + __expf(-(xr + hr)));
            const float z  = 1.0f / (1.0f + __expf(-(xz + hz)));
            const float nn = tanhf(xn + r * hn);
            const float hv = (1.0f - z) * nn + z * hreg[l];
            hreg[l] = hv;
            hdst[(size_t)b * HID + c] = (bf16)hv;
            if (ybf) ybf[row * HID + c] = (bf16)hv;
            if (yf)  yf[row * HID + c]  = hv;
            if (t == T_STEPS - 1) hn_out[(size_t)b * HID + c] = hv;
        }

        // ---- device-scope barrier: publish hdst to all 16 WGs ----
        __threadfence();
        __syncthreads();
        if (tid == 0) {
            __hip_atomic_fetch_add(ctr, 1u, __ATOMIC_ACQ_REL, __HIP_MEMORY_SCOPE_AGENT);
            const unsigned target = (unsigned)(t + 1) * NWG;
            while (__hip_atomic_load(ctr, __ATOMIC_ACQUIRE, __HIP_MEMORY_SCOPE_AGENT) < target)
                __builtin_amdgcn_s_sleep(1);
        }
        __syncthreads();
    }
}

// ---------------------------------------------------------------------------
extern "C" void kernel_launch(void* const* d_in, const int* in_sizes, int n_in,
                              void* d_out, int out_size, void* d_ws, size_t ws_size,
                              hipStream_t stream) {
    const float* x     = (const float*)d_in[0];
    const float* W_ih0 = (const float*)d_in[1];
    const float* W_hh0 = (const float*)d_in[2];
    const float* b_ih0 = (const float*)d_in[3];
    const float* b_hh0 = (const float*)d_in[4];
    const float* W_ih1 = (const float*)d_in[5];
    const float* W_hh1 = (const float*)d_in[6];
    const float* b_ih1 = (const float*)d_in[7];
    const float* b_hh1 = (const float*)d_in[8];
    float* out = (float*)d_out;

    // workspace layout (all 256B aligned)
    char* ws = (char*)d_ws;
    size_t off = 0;
    bf16* xb    = (bf16*)(ws + off); off += (size_t)T_STEPS * BATCH * DIN * 2;  // 16 MB
    bf16* bwih0 = (bf16*)(ws + off); off += (size_t)N3 * DIN * 2;               // 0.75 MB
    bf16* bwhh0 = (bf16*)(ws + off); off += (size_t)N3 * HID * 2;               // 1.5 MB
    bf16* bwih1 = (bf16*)(ws + off); off += (size_t)N3 * HID * 2;
    bf16* bwhh1 = (bf16*)(ws + off); off += (size_t)N3 * HID * 2;
    bf16* y0b   = (bf16*)(ws + off); off += (size_t)T_STEPS * BATCH * HID * 2;  // 32 MB
    float* xg   = (float*)(ws + off); off += (size_t)T_STEPS * BATCH * N3 * 4;  // 192 MB
    bf16* hb0   = (bf16*)(ws + off); off += (size_t)2 * BATCH * HID * 2;        // 128 KB
    bf16* hb1   = (bf16*)(ws + off); off += (size_t)2 * BATCH * HID * 2;
    unsigned* ctrs = (unsigned*)(ws + off); off += 256;

    gru_init<<<2048, 256, 0, stream>>>(x, W_ih0, W_hh0, W_ih1, W_hh1,
                                       xb, bwih0, bwhh0, bwih1, bwhh1,
                                       hb0, hb1, ctrs);

    const size_t y1_off = (size_t)T_STEPS * BATCH * HID;

    // Layer 0
    gru_xproj<<<dim3(N3 / 128, (T_STEPS * BATCH) / 64), 256, 0, stream>>>(
        xb, bwih0, b_ih0, xg, DIN);
    gru_recur<<<NWG, 256, 0, stream>>>(
        xg, bwhh0, b_hh0, hb0, y0b, nullptr,
        out + y1_off /* hn layer0 */, ctrs + 0);

    // Layer 1
    gru_xproj<<<dim3(N3 / 128, (T_STEPS * BATCH) / 64), 256, 0, stream>>>(
        y0b, bwih1, b_ih1, xg, HID);
    gru_recur<<<NWG, 256, 0, stream>>>(
        xg, bwhh1, b_hh1, hb1, nullptr, out /* y1 fp32 */,
        out + y1_off + (size_t)BATCH * HID /* hn layer1 */, ctrs + 1);
}
